// Hierarchical123GNN_10797547782339
// MI455X (gfx1250) — matrix operand order repaired and checked by reading, not compiled, not run on hardware
//
#include <hip/hip_runtime.h>
#include <hip/hip_bf16.h>

// ---------------- problem constants ----------------
#define NN    80      // nodes
#define DIN   64
#define HID   128
#define DOUT  16
#define PP    3160    // n*(n-1)/2 pairs
#define PPAD  3168    // pairs padded to multiple of 16
#define TT    82160   // C(80,3) triples (multiple of 16)
#define KN    96      // node dim padded to mult of 32 (K for adjacency GEMMs)
#define K2    288     // 2H+1=257 padded
#define K3    416     // 3H+4=388 padded
#define LDSPAD 8      // halves of padding per LDS B row (keeps 16B align, spreads banks)

typedef __attribute__((ext_vector_type(16))) _Float16 v16h;
typedef __attribute__((ext_vector_type(8)))  float    v8f;
typedef unsigned int u32x4 __attribute__((ext_vector_type(4)));
typedef int          i32x4 __attribute__((ext_vector_type(4)));
typedef int          i32x8 __attribute__((ext_vector_type(8)));

// ---------------------------------------------------------------------------
// Generic batched WMMA GEMM:  C[b] = A[b] (MxK, row major, f16)
//                                  x BT[b]^T (BT is NxK row major, f16)
// B panel (64 rows x K) is staged into LDS once per block via the Tensor Data
// Mover (TDM descriptor with per-row iteration -> padded LDS rows), then all
// 4 waves feed v_wmma from ds_load_b128.
// Optional fp32 AUX add (ld = ldc, unbatched), optional ReLU,
// outputs optionally fp32 and/or fp16.
// Requirements: K % 32 == 0, M % 16 == 0, N multiple of 64.
// Block = 128 threads = 4 waves; wave computes a 16x64 tile (4 accumulators).
// ---------------------------------------------------------------------------
__global__ __launch_bounds__(128) void wmma_gemm_f16(
    const _Float16* __restrict__ A, int lda, long long strideA,
    const _Float16* __restrict__ BT, int ldb, long long strideB,
    const float* __restrict__ AUX,
    float* __restrict__ Cf32, _Float16* __restrict__ Cf16,
    int ldc, long long strideC,
    int M, int N, int K, int relu)
{
  __shared__ __align__(16) _Float16 sB[64 * (K3 + LDSPAD)];

  const int wave = threadIdx.x >> 5;
  const int lane = threadIdx.x & 31;
  const long long b = blockIdx.z;
  const int m0 = blockIdx.y * 64 + wave * 16;
  const int n0 = blockIdx.x * 64;
  const int lh = lane & 15;   // lane within half-wave
  const int hi = lane >> 4;   // which half-wave
  const int ldsb = K + LDSPAD;

  const _Float16* BTb = BT + b * strideB + (long long)n0 * ldb;

#if __has_builtin(__builtin_amdgcn_tensor_load_to_lds)
  if (wave == 0) {
    // ---- Tensor DMA descriptor (D#), 2D tile iterated row-by-row ----
    unsigned ldsOff = (unsigned)(unsigned long long)(const void*)&sB[0];
    unsigned long long ga = (unsigned long long)BTb;
    u32x4 g0;
    g0[0] = 1u;                                   // count=1 (valid descriptor)
    g0[1] = ldsOff;                               // lds_addr (bytes)
    g0[2] = (unsigned)ga;                         // global_addr[95:64]
    g0[3] = (unsigned)(ga >> 32) | (2u << 30);    // global_addr hi | type=2
    i32x8 g1;
    g1[0] = (int)((1u << 16) | (1u << 19));       // data_size=2B | iterate_enable
    g1[1] = (int)(((unsigned)K & 0xFFFFu) << 16); // tensor_dim0[15:0] @ bit48
    g1[2] = (int)(((unsigned)K >> 16) | (64u << 16)); // dim0 hi | tensor_dim1=64
    g1[3] = (int)((unsigned)K << 16);             // tile_dim0=K @ bit112
    g1[4] = 1;                                    // tile_dim1=1 (one row/iter)
    g1[5] = ldb;                                  // tensor_dim0_stride (elems)
    g1[6] = 0; g1[7] = 0;                         // dim1 stride unused
    i32x4 g2;
    g2[0] = 0;                                    // tensor_dim2
    g2[1] = ldsb;                                 // lds_addr_increment (elems)
    g2[2] = ldb;                                  // global_addr_increment (elems)
    g2[3] = (63 << 16);                           // iterate_count: 64 rows
    i32x4 g3; g3[0] = 0; g3[1] = 0; g3[2] = 0; g3[3] = 0;
#if __clang_major__ >= 23
    i32x8 g4; g4[0]=0; g4[1]=0; g4[2]=0; g4[3]=0; g4[4]=0; g4[5]=0; g4[6]=0; g4[7]=0;
    __builtin_amdgcn_tensor_load_to_lds(g0, g1, g2, g3, g4, 0);
#else
    __builtin_amdgcn_tensor_load_to_lds(g0, g1, g2, g3, 0);
#endif
    __builtin_amdgcn_s_wait_tensorcnt(0);
  }
#else
  {  // cooperative fallback staging
    int chunks = K >> 3;
    for (int idx = threadIdx.x; idx < 64 * chunks; idx += 128) {
      int r = idx / chunks, c = idx % chunks;
      *(float4*)(sB + (long long)r * ldsb + c * 8) =
          *(const float4*)(BTb + (long long)r * ldb + c * 8);
    }
  }
#endif
  __syncthreads();
  if (m0 >= M) return;

  const _Float16* Ab = A + b * strideA;
  v8f acc[4] = {};

  // A fragment: lane<16 holds row m0+lh halves [k..k+7, k+16..k+23]
  //             lane>=16 holds              [k+8..k+15, k+24..k+31]
  const _Float16* arow = Ab + (long long)(m0 + lh) * lda + hi * 8;
  // B fragment from LDS: lane<16 -> local row j*16+lh, K [k..k+7, k+16..k+23]
  //                      lane>=16 ->                  K [k+8..k+15, k+24..k+31]
  const _Float16* bbase = sB + (long long)lh * ldsb + hi * 8;

  for (int k = 0; k < K; k += 32) {
    v16h afrag;
    *(float4*)&afrag         = *(const float4*)(arow + k);
    *(((float4*)&afrag) + 1) = *(const float4*)(arow + k + 16);
    __builtin_prefetch(arow + k + 64, 0, 0);
    v16h bfr[4];
#pragma unroll
    for (int j = 0; j < 4; ++j) {
      const _Float16* brow = bbase + j * 16 * ldsb + k;
      *(float4*)&bfr[j]         = *(const float4*)(brow);
      *(((float4*)&bfr[j]) + 1) = *(const float4*)(brow + 16);
    }
#pragma unroll
    for (int j = 0; j < 4; ++j)
      acc[j] = __builtin_amdgcn_wmma_f32_16x16x32_f16(
          false, afrag, false, bfr[j], (short)0, acc[j], false, false);
  }

  // C/D layout: lane<16: col n0+j*16+lh, VGPR r -> row m0+r ; lane>=16 rows +8
  const int rowb = m0 + (hi ? 8 : 0);
#pragma unroll
  for (int j = 0; j < 4; ++j) {
    const int col = n0 + j * 16 + lh;
#pragma unroll
    for (int r = 0; r < 8; ++r) {
      float v = acc[j][r];
      const long long idx = (long long)(rowb + r) * ldc + col;
      if (AUX)  v += AUX[idx];
      if (relu) v = v > 0.f ? v : 0.f;
      if (Cf32) Cf32[b * strideC + idx] = v;
      if (Cf16) Cf16[b * strideC + idx] = (_Float16)v;
    }
  }
}

// ---------------- helpers ----------------
__global__ void zero_chunks(int4* p, long long n16) {
  long long i = (long long)blockIdx.x * blockDim.x + threadIdx.x;
  if (i < n16) { int4 z = {0, 0, 0, 0}; p[i] = z; }
}

// dst[C, Rpad] (f16) = transpose(src[R, C] f32), zero pad
__global__ void transpose_f32_f16(const float* __restrict__ src,
                                  _Float16* __restrict__ dst,
                                  int R, int C, int Rpad) {
  int idx = blockIdx.x * blockDim.x + threadIdx.x;
  if (idx >= C * Rpad) return;
  int c = idx / Rpad, r = idx % Rpad;
  dst[idx] = (r < R) ? (_Float16)src[(long long)r * C + c] : (_Float16)0.f;
}

// dst[rows, colsPad] (f16) = src[rows, cols] f32, zero pad cols
__global__ void convert_pad_f16(const float* __restrict__ src,
                                _Float16* __restrict__ dst,
                                long long rows, int cols, int colsPad) {
  long long idx = (long long)blockIdx.x * blockDim.x + threadIdx.x;
  if (idx >= rows * colsPad) return;
  long long r = idx / colsPad; int c = (int)(idx % colsPad);
  dst[idx] = (c < cols) ? (_Float16)src[r * cols + c] : (_Float16)0.f;
}

__device__ __forceinline__ int pair_id(int u, int v) {
  return u * (2 * NN - u - 1) / 2 + (v - u - 1);
}

__global__ void build_structs(int* iu, int* iv, int* ta, int* tb, int* tc,
                              int* pab, int* pac, int* pbc) {
  int idx = blockIdx.x * blockDim.x + threadIdx.x;
  if (idx < PP) {
    int u = 0, rem = idx;
    while (rem >= NN - 1 - u) { rem -= NN - 1 - u; ++u; }
    iu[idx] = u; iv[idx] = u + 1 + rem;
  }
  if (idx < TT) {
    int a = 0, rem = idx;
    for (;;) {
      int c2 = (NN - 1 - a) * (NN - 2 - a) / 2;
      if (rem < c2) break;
      rem -= c2; ++a;
    }
    int m = NN - 1 - a, bb = 0;
    while (rem >= m - 1 - bb) { rem -= m - 1 - bb; ++bb; }
    int cc = bb + 1 + rem;
    int B = a + 1 + bb, C = a + 1 + cc;
    ta[idx] = a; tb[idx] = B; tc[idx] = C;
    pab[idx] = pair_id(a, B); pac[idx] = pair_id(a, C); pbc[idx] = pair_id(B, C);
  }
}

__global__ void build_adj(const int* __restrict__ ei, int E,
                          float* adjF32, _Float16* adjF16) {
  int i = blockIdx.x * blockDim.x + threadIdx.x;
  if (i >= E) return;
  int s = ei[i], d = ei[E + i];
  if (s == d) return;
  adjF32[s * NN + d] = 1.f; adjF32[d * NN + s] = 1.f;
  adjF16[s * KN + d] = (_Float16)1.f; adjF16[d * KN + s] = (_Float16)1.f;
}

__global__ void build_z2(const _Float16* __restrict__ h, const float* __restrict__ adj,
                         const int* __restrict__ iu, const int* __restrict__ iv,
                         _Float16* __restrict__ z2) {
  int idx = blockIdx.x * blockDim.x + threadIdx.x;
  if (idx >= PPAD * K2) return;
  int p = idx / K2, c = idx % K2;
  _Float16 v = (_Float16)0.f;
  if (p < PP) {
    int u = iu[p], w = iv[p];
    if (c < HID)            v = h[u * HID + c];
    else if (c < 2 * HID)   v = h[w * HID + (c - HID)];
    else if (c == 2 * HID)  v = (_Float16)adj[u * NN + w];
  }
  z2[idx] = v;
}

__global__ void build_z3(const _Float16* __restrict__ h, const float* __restrict__ adj,
                         const int* __restrict__ ta, const int* __restrict__ tb,
                         const int* __restrict__ tc, _Float16* __restrict__ z3) {
  long long idx = (long long)blockIdx.x * blockDim.x + threadIdx.x;
  if (idx >= (long long)TT * K3) return;
  int t = (int)(idx / K3), c = (int)(idx % K3);
  int a = ta[t], b = tb[t], cc = tc[t];
  _Float16 v = (_Float16)0.f;
  if (c < HID)            v = h[a * HID + c];
  else if (c < 2 * HID)   v = h[b * HID + (c - HID)];
  else if (c < 3 * HID)   v = h[cc * HID + (c - 2 * HID)];
  else if (c < 3 * HID + 4) {
    int e = (int)(adj[a * NN + b] + adj[a * NN + cc] + adj[b * NN + cc] + 0.5f);
    v = (c - 3 * HID == e) ? (_Float16)1.f : (_Float16)0.f;
  }
  z3[idx] = v;
}

// Md2t[s][h][o] = message of pair {o,s}
__global__ void scatter2(const _Float16* __restrict__ m2, const int* __restrict__ iu,
                         const int* __restrict__ iv, _Float16* __restrict__ Md2t) {
  int idx = blockIdx.x * blockDim.x + threadIdx.x;
  if (idx >= PP * HID) return;
  int p = idx / HID, hh = idx % HID;
  _Float16 v = m2[(long long)p * HID + hh];
  int u = iu[p], w = iv[p];
  Md2t[((long long)w * HID + hh) * KN + u] = v;
  Md2t[((long long)u * HID + hh) * KN + w] = v;
}

__global__ void gather2(const _Float16* __restrict__ Nd2, const float* __restrict__ adj,
                        const int* __restrict__ iu, const int* __restrict__ iv,
                        float* __restrict__ aux2) {
  int idx = blockIdx.x * blockDim.x + threadIdx.x;
  if (idx >= PPAD * HID) return;
  int p = idx / HID, hh = idx % HID;
  float v = 0.f;
  if (p < PP) {
    int u = iu[p], w = iv[p];
    v = adj[u * NN + w] * ((float)Nd2[((long long)u * NN + w) * HID + hh] +
                           (float)Nd2[((long long)w * NN + u) * HID + hh]);
  }
  aux2[idx] = v;
}

// Md3t[pair][h][third] : conflict-free scatter (each cell sourced by one triple)
__global__ void scatter3(const _Float16* __restrict__ m3,
                         const int* __restrict__ ta, const int* __restrict__ tb,
                         const int* __restrict__ tc, const int* __restrict__ pab,
                         const int* __restrict__ pac, const int* __restrict__ pbc,
                         _Float16* __restrict__ Md3t) {
  long long idx = (long long)blockIdx.x * blockDim.x + threadIdx.x;
  if (idx >= (long long)TT * HID) return;
  int t = (int)(idx / HID), hh = (int)(idx % HID);
  _Float16 v = m3[idx];
  Md3t[((long long)pab[t] * HID + hh) * KN + tc[t]] = v;
  Md3t[((long long)pac[t] * HID + hh) * KN + tb[t]] = v;
  Md3t[((long long)pbc[t] * HID + hh) * KN + ta[t]] = v;
}

__global__ void gather3(const _Float16* __restrict__ Nd3,
                        const int* __restrict__ ta, const int* __restrict__ tb,
                        const int* __restrict__ tc, const int* __restrict__ pab,
                        const int* __restrict__ pac, const int* __restrict__ pbc,
                        float* __restrict__ aux3) {
  long long idx = (long long)blockIdx.x * blockDim.x + threadIdx.x;
  if (idx >= (long long)TT * HID) return;
  int t = (int)(idx / HID), hh = (int)(idx % HID);
  aux3[idx] = (float)Nd3[((long long)pab[t] * NN + tc[t]) * HID + hh] +
              (float)Nd3[((long long)pac[t] * NN + tb[t]) * HID + hh] +
              (float)Nd3[((long long)pbc[t] * NN + ta[t]) * HID + hh];
}

__global__ void reduce_rows(const _Float16* __restrict__ X, long long rows,
                            float* __restrict__ out) {
  int c = threadIdx.x; // HID threads
  float s = 0.f;
  for (long long r = blockIdx.x; r < rows; r += gridDim.x)
    s += (float)X[r * HID + c];
  atomicAdd(&out[c], s);
}

__global__ void classifier(const float* __restrict__ g, const float* __restrict__ w1,
                           const float* __restrict__ b1, const float* __restrict__ w2,
                           const float* __restrict__ b2, float* __restrict__ out) {
  __shared__ float zs[HID];
  int j = threadIdx.x;
  float s = b1[j];
  for (int k = 0; k < 3 * HID; ++k) s += g[k] * w1[k * HID + j];
  zs[j] = s > 0.f ? s : 0.f;
  __syncthreads();
  if (j < DOUT) {
    float o = b2[j];
    for (int k = 0; k < HID; ++k) o += zs[k] * w2[k * DOUT + j];
    out[j] = o;
  }
}

// ---------------- host-side orchestration ----------------
static inline size_t al256(size_t x) { return (x + 255) & ~(size_t)255; }

static void gemm(hipStream_t st, const _Float16* A, int lda, long long sA,
                 const _Float16* BT, int ldb, long long sB, const float* AUX,
                 float* Cf, _Float16* Ch, int ldc, long long sC,
                 int M, int N, int K, int relu, int batches) {
  dim3 grid((N + 63) / 64, (M + 63) / 64, batches);
  wmma_gemm_f16<<<grid, 128, 0, st>>>(A, lda, sA, BT, ldb, sB, AUX, Cf, Ch,
                                      ldc, sC, M, N, K, relu);
}
static void zero_(hipStream_t st, void* p, size_t bytes) {
  long long n16 = (long long)(bytes / 16);
  zero_chunks<<<(unsigned)((n16 + 255) / 256), 256, 0, st>>>((int4*)p, n16);
}

extern "C" void kernel_launch(void* const* d_in, const int* in_sizes, int n_in,
                              void* d_out, int out_size, void* d_ws, size_t ws_size,
                              hipStream_t stream) {
  // input order (setup_inputs): x, g1_w1[0..2], g1_w2[0..2], g2_w1[0..1],
  // g2_w2[0..1], g3_w1[0..1], g3_w2[0..1], c_w1, c_b1, c_w2, c_b2, edge_index, batch
  const float* x      = (const float*)d_in[0];
  const float* g1w1[3] = {(const float*)d_in[1], (const float*)d_in[2], (const float*)d_in[3]};
  const float* g1w2[3] = {(const float*)d_in[4], (const float*)d_in[5], (const float*)d_in[6]};
  const float* g2w1[2] = {(const float*)d_in[7], (const float*)d_in[8]};
  const float* g2w2[2] = {(const float*)d_in[9], (const float*)d_in[10]};
  const float* g3w1[2] = {(const float*)d_in[11], (const float*)d_in[12]};
  const float* g3w2[2] = {(const float*)d_in[13], (const float*)d_in[14]};
  const float* cw1 = (const float*)d_in[15];
  const float* cb1 = (const float*)d_in[16];
  const float* cw2 = (const float*)d_in[17];
  const float* cb2 = (const float*)d_in[18];
  const int*   ei  = (const int*)d_in[19];
  const int    E   = in_sizes[19] / 2;
  float* out = (float*)d_out;

  // -------- workspace arena --------
  char* w = (char*)d_ws;
  auto alloc = [&](size_t bytes) -> void* { void* p = (void*)w; w += al256(bytes); return p; };

  int* iu  = (int*)alloc(PP * 4);   int* iv  = (int*)alloc(PP * 4);
  int* ta  = (int*)alloc(TT * 4);   int* tb  = (int*)alloc(TT * 4);
  int* tc  = (int*)alloc(TT * 4);
  int* pab = (int*)alloc(TT * 4);   int* pac = (int*)alloc(TT * 4);
  int* pbc = (int*)alloc(TT * 4);

  float*     adjF32 = (float*)alloc(NN * NN * 4);
  _Float16*  adjF16 = (_Float16*)alloc(NN * KN * 2);
  _Float16*  xF16   = (_Float16*)alloc(NN * DIN * 2);

  _Float16 *w1t1[3], *w2t1[3], *w1t2[2], *w2t2[2], *w1t3[2], *w2t3[2];
  int k1[3] = {DIN, HID, HID};
  for (int l = 0; l < 3; ++l) { w1t1[l] = (_Float16*)alloc(HID * k1[l] * 2);
                                w2t1[l] = (_Float16*)alloc(HID * k1[l] * 2); }
  int kk2[2] = {K2, HID};
  for (int l = 0; l < 2; ++l) { w1t2[l] = (_Float16*)alloc(HID * kk2[l] * 2);
                                w2t2[l] = (_Float16*)alloc(HID * kk2[l] * 2); }
  int kk3[2] = {K3, HID};
  for (int l = 0; l < 2; ++l) { w1t3[l] = (_Float16*)alloc(HID * kk3[l] * 2);
                                w2t3[l] = (_Float16*)alloc(HID * kk3[l] * 2); }

  float*    t_f32 = (float*)alloc(NN * HID * 4);
  _Float16* tT    = (_Float16*)alloc(HID * KN * 2);
  float*    u_f32 = (float*)alloc(NN * HID * 4);
  _Float16* hF16  = (_Float16*)alloc(NN * HID * 2);

  _Float16* z2a  = (_Float16*)alloc((size_t)PPAD * K2 * 2);
  _Float16* z2b  = (_Float16*)alloc((size_t)PPAD * HID * 2);
  _Float16* m2   = (_Float16*)alloc((size_t)PPAD * HID * 2);
  _Float16* Md2t = (_Float16*)alloc((size_t)NN * HID * KN * 2);
  _Float16* Nd2  = (_Float16*)alloc((size_t)NN * NN * HID * 2);
  float*    aux2 = (float*)alloc((size_t)PPAD * HID * 4);

  _Float16* z3a  = (_Float16*)alloc((size_t)TT * K3 * 2);
  _Float16* z3b  = (_Float16*)alloc((size_t)TT * HID * 2);
  _Float16* m3   = (_Float16*)alloc((size_t)TT * HID * 2);
  _Float16* Md3t = (_Float16*)alloc((size_t)PP * HID * KN * 2);   // 77.6 MB
  _Float16* Nd3  = (_Float16*)alloc((size_t)PP * NN * HID * 2);   // 64.7 MB
  float*    aux3 = (float*)Md3t;  // alias: Md3t dead once Nd3 is computed
  float*    gpool = (float*)alloc(3 * HID * 4);
  (void)ws_size;

  // -------- setup --------
  build_structs<<<(TT + 255) / 256, 256, 0, stream>>>(iu, iv, ta, tb, tc, pab, pac, pbc);
  zero_(stream, adjF32, NN * NN * 4);
  zero_(stream, adjF16, NN * KN * 2);
  build_adj<<<(E + 255) / 256, 256, 0, stream>>>(ei, E, adjF32, adjF16);
  convert_pad_f16<<<(NN * DIN + 255) / 256, 256, 0, stream>>>(x, xF16, NN, DIN, DIN);

  for (int l = 0; l < 3; ++l) {
    int nT = HID * k1[l];
    transpose_f32_f16<<<(nT + 255) / 256, 256, 0, stream>>>(g1w1[l], w1t1[l], k1[l], HID, k1[l]);
    transpose_f32_f16<<<(nT + 255) / 256, 256, 0, stream>>>(g1w2[l], w2t1[l], k1[l], HID, k1[l]);
  }
  int kr2[2] = {2 * HID + 1, HID};
  for (int l = 0; l < 2; ++l) {
    int nT = HID * kk2[l];
    transpose_f32_f16<<<(nT + 255) / 256, 256, 0, stream>>>(g2w1[l], w1t2[l], kr2[l], HID, kk2[l]);
    transpose_f32_f16<<<(nT + 255) / 256, 256, 0, stream>>>(g2w2[l], w2t2[l], kr2[l], HID, kk2[l]);
  }
  int kr3[2] = {3 * HID + 4, HID};
  for (int l = 0; l < 2; ++l) {
    int nT = HID * kk3[l];
    transpose_f32_f16<<<(nT + 255) / 256, 256, 0, stream>>>(g3w1[l], w1t3[l], kr3[l], HID, kk3[l]);
    transpose_f32_f16<<<(nT + 255) / 256, 256, 0, stream>>>(g3w2[l], w2t3[l], kr3[l], HID, kk3[l]);
  }

  // -------- stage 1: 1-GNN (3 layers) --------
  for (int l = 0; l < 3; ++l) {
    const _Float16* Ain = (l == 0) ? xF16 : hF16;
    int K = k1[l];
    // t = h @ W2
    gemm(stream, Ain, K, 0, w2t1[l], K, 0, nullptr, t_f32, nullptr, HID, 0, NN, HID, K, 0, 1);
    transpose_f32_f16<<<(HID * KN + 255) / 256, 256, 0, stream>>>(t_f32, tT, NN, HID, KN);
    // u = h @ W1
    gemm(stream, Ain, K, 0, w1t1[l], K, 0, nullptr, u_f32, nullptr, HID, 0, NN, HID, K, 0, 1);
    // h = relu(u + adj @ t)
    gemm(stream, adjF16, KN, 0, tT, KN, 0, u_f32, nullptr, hF16, HID, 0, NN, HID, KN, 1, 1);
  }

  // -------- stage 2: 2-GNN (2 layers) --------
  build_z2<<<(PPAD * K2 + 255) / 256, 256, 0, stream>>>(hF16, adjF32, iu, iv, z2a);
  for (int l = 0; l < 2; ++l) {
    const _Float16* Ain = (l == 0) ? z2a : z2b;
    _Float16* Aout      = (l == 0) ? z2b : z2a;   // reuse z2a region as [PPAD,128]
    int K = kk2[l];
    gemm(stream, Ain, K, 0, w2t2[l], K, 0, nullptr, nullptr, m2, HID, 0, PPAD, HID, K, 0, 1);
    zero_(stream, Md2t, (size_t)NN * HID * KN * 2);
    scatter2<<<(PP * HID + 255) / 256, 256, 0, stream>>>(m2, iu, iv, Md2t);
    gemm(stream, adjF16, KN, 0, Md2t, KN, (long long)HID * KN, nullptr,
         nullptr, Nd2, HID, (long long)NN * HID, NN, HID, KN, 0, NN);
    gather2<<<(PPAD * HID + 255) / 256, 256, 0, stream>>>(Nd2, adjF32, iu, iv, aux2);
    gemm(stream, Ain, K, 0, w1t2[l], K, 0, aux2, nullptr, Aout, HID, 0, PPAD, HID, K, 1, 1);
  }

  // -------- stage 3: 3-GNN (2 layers) --------
  build_z3<<<(unsigned)(((long long)TT * K3 + 255) / 256), 256, 0, stream>>>(hF16, adjF32, ta, tb, tc, z3a);
  for (int l = 0; l < 2; ++l) {
    const _Float16* Ain = (l == 0) ? z3a : z3b;
    _Float16* Aout      = (l == 0) ? z3b : z3a;   // reuse z3a region as [TT,128]
    int K = kk3[l];
    gemm(stream, Ain, K, 0, w2t3[l], K, 0, nullptr, nullptr, m3, HID, 0, TT, HID, K, 0, 1);
    zero_(stream, Md3t, (size_t)PP * HID * KN * 2);
    scatter3<<<(unsigned)(((long long)TT * HID + 255) / 256), 256, 0, stream>>>(
        m3, ta, tb, tc, pab, pac, pbc, Md3t);
    gemm(stream, adjF16, KN, 0, Md3t, KN, (long long)HID * KN, nullptr,
         nullptr, Nd3, HID, (long long)NN * HID, NN, HID, KN, 0, PP);
    gather3<<<(unsigned)(((long long)TT * HID + 255) / 256), 256, 0, stream>>>(
        Nd3, ta, tb, tc, pab, pac, pbc, aux3);
    gemm(stream, Ain, K, 0, w1t3[l], K, 0, aux3, nullptr, Aout, HID, 0, TT, HID, K, 1, 1);
  }

  // -------- pooling + classifier --------
  zero_(stream, gpool, 3 * HID * 4);
  reduce_rows<<<64, HID, 0, stream>>>(hF16, NN, gpool);
  reduce_rows<<<128, HID, 0, stream>>>(z2a, PPAD, gpool + HID);       // final z2
  reduce_rows<<<512, HID, 0, stream>>>(z3a, TT, gpool + 2 * HID);     // final z3
  classifier<<<1, HID, 0, stream>>>(gpool, cw1, cb1, cw2, cb2, out);
  (void)out_size; (void)n_in;
}
